// MambaFlowBlock_4913442586863
// MI455X (gfx1250) — compile-verified, matrix operand order checked by reading, not run on hardware
//
#include <hip/hip_runtime.h>
#include <hip/hip_bf16.h>
#include <math.h>

// ---- problem dims ----
#define D_   512
#define NH_  8
#define HD_  64
#define DFF_ 2048
#define TD_  512
#define NST_ 16
#define DC_  4
#define ED_  1024
#define DR_  32
#define B_   2
#define T_   4096
#define TC_  1024

typedef __attribute__((ext_vector_type(16))) __bf16 v16bf;
typedef __attribute__((ext_vector_type(8)))  float  v8f;

union BF16Frag { v16bf v; unsigned short u[16]; };
union F32Acc   { v8f   v; float f[8]; };

__device__ __forceinline__ unsigned short f2bf(float f) {
  union { float f; unsigned u; } x; x.f = f;
  unsigned r = x.u + 0x7FFFu + ((x.u >> 16) & 1u);   // round-to-nearest-even
  return (unsigned short)(r >> 16);
}

__device__ __forceinline__ float act_apply(float x, int act) {
  if (act == 1) return x * (1.f / (1.f + __expf(-x)));            // silu
  if (act == 2) return 0.5f * x * (1.f + erff(x * 0.70710678f));  // exact gelu
  if (act == 3) return (x > 20.f) ? x : log1pf(__expf(x));        // softplus
  return x;
}

// ---------------------------------------------------------------------------
// Generic strided GEMM: C[m*ldc+n] (= or +=) act( sum_k A[m*lda+k]*W[k*wsk+n*wsn] + bias[n] )
// Block = 128 threads (4 waves), 64x64 block tile, each wave a 32x32 register
// tile -> 4x v_wmma_f32_16x16x32_bf16 per 32-wide K step.
//
// LDS tiles are stored in *fragment order*: sXf[sub16][lane][16 bf16], so each
// lane reads its entire WMMA operand fragment as one contiguous 32B LDS load
// (2x ds_load_b128) instead of 64 scalar ds_load_u16.
// Swizzle invariant: a float4 along K (k0 % 4 == 0, within an 8-aligned K
// half-block) maps to 4 consecutive fragment elements of one lane:
//   lane = (row_or_col & 15) + 16*((k>>3)&1)
//   i0   = 8*((k>>4)&1) + 2*((k>>1)&3)
// so staging is: global_load_b128 -> cvt -> single ds_store_b64.
// Double-buffered LDS: one barrier per K iteration.
// Assumes K % 32 == 0 (true for every launch in this workload).
// ---------------------------------------------------------------------------
__global__ void __launch_bounds__(128)
gemm_bf16_wmma(const float* __restrict__ A, int lda,
               const float* __restrict__ W, int wsk, int wsn,
               const float* __restrict__ bias,
               float* __restrict__ C, int ldc,
               int M, int N, int K, int act, int accum)
{
  __shared__ __align__(32) unsigned short sAf[2][4][32][16];  // 8 KB
  __shared__ __align__(32) unsigned short sBf[2][4][32][16];  // 8 KB

  const int tid  = threadIdx.x;
  const int lane = tid & 31;
  const int wave = tid >> 5;
  const int wm = (wave >> 1) * 32;
  const int wn = (wave & 1) * 32;
  const int lr = lane & 15;
  const int kh = lane >> 4;
  const int m0 = blockIdx.y * 64;
  const int n0 = blockIdx.x * 64;

  F32Acc acc[2][2];
  #pragma unroll
  for (int p = 0; p < 2; ++p)
    #pragma unroll
    for (int q = 0; q < 2; ++q)
      #pragma unroll
      for (int r = 0; r < 8; ++r) acc[p][q].f[r] = 0.f;

  // stage one 64x32 A tile + 32x64 W tile (bf16, fragment-ordered) into buffer `bi`
  auto stage = [&](int kt, int bi) {
    // A: 512 float4 quads, each -> one ds_store_b64
    #pragma unroll
    for (int j = 0; j < 4; ++j) {
      int idx = tid + j * 128;          // 0..511
      int r   = idx >> 3;               // 0..63
      int c4  = (idx & 7) * 4;          // 0..28
      int gm  = m0 + r;
      float4 v = make_float4(0.f, 0.f, 0.f, 0.f);
      if (gm < M) v = *(const float4*)(A + (size_t)gm * lda + kt + c4);
      int s    = r >> 4;
      int ln   = (r & 15) + (((c4 >> 3) & 1) << 4);
      int i0   = (((c4 >> 4) & 1) << 3) + (((c4 >> 1) & 3) << 1);
      *(ushort4*)&sAf[bi][s][ln][i0] = make_ushort4(f2bf(v.x), f2bf(v.y), f2bf(v.z), f2bf(v.w));
    }
    // W: 512 k-quads (4 consecutive k, fixed n), each -> one ds_store_b64
    #pragma unroll
    for (int j = 0; j < 4; ++j) {
      int idx = tid + j * 128;          // 0..511
      int n   = idx & 63;
      int k0  = (idx >> 6) * 4;         // 0..28
      int gn  = n0 + n;
      float4 v = make_float4(0.f, 0.f, 0.f, 0.f);
      if (gn < N) {
        if (wsk == 1) {                 // k-contiguous (e.g. q @ k^T): true b128 load
          v = *(const float4*)(W + (size_t)(kt + k0) + (size_t)gn * wsn);
        } else {                        // row-major weights: 4 strided dwords
          const float* wp = W + (size_t)(kt + k0) * wsk + (size_t)gn * wsn;
          v.x = wp[0]; v.y = wp[wsk]; v.z = wp[2 * (size_t)wsk]; v.w = wp[3 * (size_t)wsk];
        }
      }
      int sn = n >> 4;
      int ln = (n & 15) + (((k0 >> 3) & 1) << 4);
      int i0 = (((k0 >> 4) & 1) << 3) + (((k0 >> 1) & 3) << 1);
      *(ushort4*)&sBf[bi][sn][ln][i0] = make_ushort4(f2bf(v.x), f2bf(v.y), f2bf(v.z), f2bf(v.w));
    }
  };

  const int nk = K >> 5;     // K % 32 == 0 for all launches here
  stage(0, 0);
  __syncthreads();

  const int sa = wm >> 4;    // 0 or 2
  const int sb = wn >> 4;

  for (int it = 0; it < nk; ++it) {
    int cur = it & 1;
    int kt  = it << 5;
    // speculative prefetch of tile it+2 (uniform condition; OOB lanes are dropped by HW)
    if (kt + 64 < K) {
      __builtin_prefetch(A + (size_t)(m0 + (tid >> 1)) * lda + kt + 64, 0, 1);
      __builtin_prefetch(W + (size_t)(kt + 64 + (tid & 31)) * wsk + (size_t)n0 * wsn, 0, 1);
    }
    if (it + 1 < nk) stage(kt + 32, cur ^ 1);

    BF16Frag a0, a1, b0, b1;
    a0.v = *(const v16bf*)&sAf[cur][sa    ][lane][0];
    a1.v = *(const v16bf*)&sAf[cur][sa + 1][lane][0];
    b0.v = *(const v16bf*)&sBf[cur][sb    ][lane][0];
    b1.v = *(const v16bf*)&sBf[cur][sb + 1][lane][0];

    acc[0][0].v = __builtin_amdgcn_wmma_f32_16x16x32_bf16(false, a0.v, false, b0.v, (short)0, acc[0][0].v, false, false);
    acc[0][1].v = __builtin_amdgcn_wmma_f32_16x16x32_bf16(false, a0.v, false, b1.v, (short)0, acc[0][1].v, false, false);
    acc[1][0].v = __builtin_amdgcn_wmma_f32_16x16x32_bf16(false, a1.v, false, b0.v, (short)0, acc[1][0].v, false, false);
    acc[1][1].v = __builtin_amdgcn_wmma_f32_16x16x32_bf16(false, a1.v, false, b1.v, (short)0, acc[1][1].v, false, false);
    __syncthreads();
  }

  // epilogue: 16x16 f32 C/D layout -> lane (l&15)=N col, vgpr v + 8*(l>>4) = M row
  #pragma unroll
  for (int p = 0; p < 2; ++p) {
    #pragma unroll
    for (int q = 0; q < 2; ++q) {
      int mb = m0 + wm + p * 16;
      int n  = n0 + wn + q * 16 + lr;
      #pragma unroll
      for (int v = 0; v < 8; ++v) {
        int m = mb + v + 8 * kh;
        if (m < M && n < N) {
          float val = acc[p][q].f[v];
          if (bias) val += bias[n];
          val = act_apply(val, act);
          size_t o = (size_t)m * ldc + n;
          if (accum) val += C[o];
          C[o] = val;
        }
      }
    }
  }
}

// ---------------------------------------------------------------------------
// mod = silu(t_emb) @ ada_w + ada_b   (M = B = 2: scalar kernel)
// ---------------------------------------------------------------------------
__global__ void k_modulation(const float* __restrict__ t_emb, const float* __restrict__ ada_w,
                             const float* __restrict__ ada_b, float* __restrict__ mod)
{
  int idx = blockIdx.x * blockDim.x + threadIdx.x;
  if (idx >= B_ * 9 * D_) return;
  int b = idx / (9 * D_), j = idx % (9 * D_);
  float s = 0.f;
  for (int k = 0; k < TD_; ++k) {
    float t = t_emb[b * TD_ + k];
    s += (t * (1.f / (1.f + __expf(-t)))) * ada_w[(size_t)k * (9 * D_) + j];
  }
  mod[idx] = s + ada_b[j];
}

// LayerNorm + AdaLN modulate: out = ln(x)*(1+scale)+shift ; one block per (b,t) row
__global__ void __launch_bounds__(256)
k_ln_mod(const float* __restrict__ x, const float* __restrict__ mod,
         int shIdx, int scIdx, float* __restrict__ out)
{
  __shared__ float red[256];
  int row = blockIdx.x;
  int b = row / T_;
  const float* xr = x + (size_t)row * D_;
  float v0 = xr[threadIdx.x], v1 = xr[threadIdx.x + 256];
  red[threadIdx.x] = v0 + v1; __syncthreads();
  for (int st = 128; st > 0; st >>= 1) { if (threadIdx.x < st) red[threadIdx.x] += red[threadIdx.x + st]; __syncthreads(); }
  float mu = red[0] * (1.f / D_);
  __syncthreads();
  float d0 = v0 - mu, d1 = v1 - mu;
  red[threadIdx.x] = d0 * d0 + d1 * d1; __syncthreads();
  for (int st = 128; st > 0; st >>= 1) { if (threadIdx.x < st) red[threadIdx.x] += red[threadIdx.x + st]; __syncthreads(); }
  float rs = rsqrtf(red[0] * (1.f / D_) + 1e-5f);
  const float* mrow = mod + (size_t)b * 9 * D_;
  float* orow = out + (size_t)row * D_;
  int d = threadIdx.x;
  orow[d] = d0 * rs * (1.f + mrow[scIdx * D_ + d]) + mrow[shIdx * D_ + d];
  d = threadIdx.x + 256;
  orow[d] = d1 * rs * (1.f + mrow[scIdx * D_ + d]) + mrow[shIdx * D_ + d];
}

__global__ void k_flip(const float* __restrict__ in, float* __restrict__ out) {
  size_t idx = (size_t)blockIdx.x * blockDim.x + threadIdx.x;
  if (idx >= (size_t)B_ * T_ * D_) return;
  int d = idx % D_; int t = (idx / D_) % T_; int b = idx / ((size_t)D_ * T_);
  out[idx] = in[((size_t)b * T_ + (T_ - 1 - t)) * D_ + d];
}

// causal depthwise conv over T + silu; xi = first ED cols of xz (row stride 2*ED)
__global__ void k_dwconv_silu(const float* __restrict__ xz, const float* __restrict__ cw,
                              const float* __restrict__ cb, float* __restrict__ xc)
{
  size_t idx = (size_t)blockIdx.x * blockDim.x + threadIdx.x;
  if (idx >= (size_t)B_ * T_ * ED_) return;
  int e = idx % ED_; int t = (idx / ED_) % T_; int b = idx / ((size_t)ED_ * T_);
  float s = cb[e];
  #pragma unroll
  for (int j = 0; j < DC_; ++j) {
    int tt = t - (DC_ - 1) + j;
    if (tt >= 0) s += xz[((size_t)b * T_ + tt) * (2 * ED_) + e] * cw[e * DC_ + j];
  }
  xc[idx] = s * (1.f / (1.f + __expf(-s)));
}

// selective-scan recurrence: one thread per (b, e), 16 states in registers
__global__ void __launch_bounds__(256)
k_scan(const float* __restrict__ delta, const float* __restrict__ u,
       const float* __restrict__ dbc, const float* __restrict__ A_log,
       const float* __restrict__ Dp, float* __restrict__ ys)
{
  int idx = blockIdx.x * 256 + threadIdx.x;
  if (idx >= B_ * ED_) return;
  int b = idx / ED_, e = idx % ED_;
  float A[NST_], h[NST_];
  #pragma unroll
  for (int n = 0; n < NST_; ++n) { A[n] = -__expf(A_log[e * NST_ + n]); h[n] = 0.f; }
  float dp = Dp[e];
  for (int t = 0; t < T_; ++t) {
    size_t r = (size_t)b * T_ + t;
    float d  = delta[r * ED_ + e];
    float uu = u[r * ED_ + e];
    float du = d * uu;
    const float* bc = dbc + r * (DR_ + 2 * NST_);
    float y = 0.f;
    #pragma unroll
    for (int n = 0; n < NST_; ++n) {
      float hn = __expf(d * A[n]) * h[n] + du * bc[DR_ + n];
      h[n] = hn;
      y += hn * bc[DR_ + NST_ + n];
    }
    ys[r * ED_ + e] = y + uu * dp;
  }
}

// ys *= silu(z),  z = second half of xz row
__global__ void k_gate(const float* __restrict__ xz, float* __restrict__ ys) {
  size_t idx = (size_t)blockIdx.x * blockDim.x + threadIdx.x;
  if (idx >= (size_t)B_ * T_ * ED_) return;
  size_t row = idx / ED_; int e = idx % ED_;
  float z = xz[row * (2 * ED_) + ED_ + e];
  ys[idx] *= z * (1.f / (1.f + __expf(-z)));
}

// out = base + gate[b,d] * val
__global__ void k_resgate(const float* __restrict__ base, const float* __restrict__ val,
                          const float* __restrict__ mod, int gIdx, float* __restrict__ out)
{
  size_t idx = (size_t)blockIdx.x * blockDim.x + threadIdx.x;
  if (idx >= (size_t)B_ * T_ * D_) return;
  int b = idx / ((size_t)T_ * D_); int d = idx % D_;
  out[idx] = base[idx] + mod[(size_t)b * 9 * D_ + gIdx * D_ + d] * val[idx];
}

// GPT-NeoX rotate-half RoPE in place; rows = B*L, head cols h*64..h*64+63
__global__ void k_rope(float* __restrict__ x, int L) {
  size_t idx = (size_t)blockIdx.x * blockDim.x + threadIdx.x;
  size_t total = (size_t)B_ * L * NH_ * 32;
  if (idx >= total) return;
  int i = idx & 31; int h = (idx >> 5) & 7; size_t row = idx >> 8;
  int t = (int)(row % L);
  float inv = __expf(-(float)i * (9.21034037f / 32.f));  // 10000^{-2i/64}
  float ang = (float)t * inv;
  float c = cosf(ang), s = sinf(ang);
  float* p = x + row * D_ + h * HD_;
  float x1 = p[i], x2 = p[i + 32];
  p[i]      = x1 * c - x2 * s;
  p[i + 32] = x2 * c + x1 * s;
}

// row softmax over TC with pre-scale; one block per row
__global__ void __launch_bounds__(256) k_softmax(float* __restrict__ S, float scale) {
  __shared__ float red[256];
  float* row = S + (size_t)blockIdx.x * TC_;
  float v[4]; float m = -1e30f;
  #pragma unroll
  for (int j = 0; j < 4; ++j) { v[j] = row[threadIdx.x + j * 256] * scale; m = fmaxf(m, v[j]); }
  red[threadIdx.x] = m; __syncthreads();
  for (int st = 128; st > 0; st >>= 1) { if (threadIdx.x < st) red[threadIdx.x] = fmaxf(red[threadIdx.x], red[threadIdx.x + st]); __syncthreads(); }
  m = red[0]; __syncthreads();
  float s = 0.f;
  #pragma unroll
  for (int j = 0; j < 4; ++j) { v[j] = __expf(v[j] - m); s += v[j]; }
  red[threadIdx.x] = s; __syncthreads();
  for (int st = 128; st > 0; st >>= 1) { if (threadIdx.x < st) red[threadIdx.x] += red[threadIdx.x + st]; __syncthreads(); }
  float inv = 1.f / red[0];
  #pragma unroll
  for (int j = 0; j < 4; ++j) row[threadIdx.x + j * 256] = v[j] * inv;
}

// ---------------------------------------------------------------------------
// host side
// ---------------------------------------------------------------------------
struct MambaP { const float *in_w, *conv_w, *conv_b, *xproj_w, *dt_w, *dt_b, *A_log, *Dp, *out_w; };

static inline void gemm(hipStream_t st, const float* A, int lda, const float* W, int wsk, int wsn,
                        const float* bias, float* C, int ldc, int M, int N, int K, int act, int accum)
{
  dim3 g((N + 63) / 64, (M + 63) / 64);
  gemm_bf16_wmma<<<g, 128, 0, st>>>(A, lda, W, wsk, wsn, bias, C, ldc, M, N, K, act, accum);
}

static void run_mamba(hipStream_t st, const float* xin, const MambaP& p,
                      float* xz, float* xconv, float* dbc, float* delta, float* ys, float* yout)
{
  const int M = B_ * T_;
  gemm(st, xin, D_, p.in_w, 2 * ED_, 1, nullptr, xz, 2 * ED_, M, 2 * ED_, D_, 0, 0);
  size_t tot = (size_t)M * ED_;
  k_dwconv_silu<<<(tot + 255) / 256, 256, 0, st>>>(xz, p.conv_w, p.conv_b, xconv);
  gemm(st, xconv, ED_, p.xproj_w, DR_ + 2 * NST_, 1, nullptr, dbc, DR_ + 2 * NST_, M, DR_ + 2 * NST_, ED_, 0, 0);
  gemm(st, dbc, DR_ + 2 * NST_, p.dt_w, ED_, 1, p.dt_b, delta, ED_, M, ED_, DR_, 3, 0); // softplus
  k_scan<<<(B_ * ED_ + 255) / 256, 256, 0, st>>>(delta, xconv, dbc, p.A_log, p.Dp, ys);
  k_gate<<<(tot + 255) / 256, 256, 0, st>>>(xz, ys);
  gemm(st, ys, ED_, p.out_w, D_, 1, nullptr, yout, D_, M, D_, ED_, 0, 0);
}

extern "C" void kernel_launch(void* const* d_in, const int* in_sizes, int n_in,
                              void* d_out, int out_size, void* d_ws, size_t ws_size,
                              hipStream_t stream)
{
  (void)in_sizes; (void)n_in; (void)out_size; (void)ws_size;
  const float* x_in  = (const float*)d_in[0];
  const float* t_emb = (const float*)d_in[1];
  const float* ctx   = (const float*)d_in[2];
  const float* ada_w = (const float*)d_in[3];
  const float* ada_b = (const float*)d_in[4];
  auto mp = [&](int base) {
    MambaP p;
    p.in_w = (const float*)d_in[base + 0]; p.conv_w = (const float*)d_in[base + 1];
    p.conv_b = (const float*)d_in[base + 2]; p.xproj_w = (const float*)d_in[base + 3];
    p.dt_w = (const float*)d_in[base + 4]; p.dt_b = (const float*)d_in[base + 5];
    p.A_log = (const float*)d_in[base + 6]; p.Dp = (const float*)d_in[base + 7];
    p.out_w = (const float*)d_in[base + 8];
    return p;
  };
  MambaP pf = mp(5), pb = mp(14);
  const float* merge_w = (const float*)d_in[23];
  const float* wq = (const float*)d_in[24]; const float* bq = (const float*)d_in[25];
  const float* wk = (const float*)d_in[26]; const float* bk = (const float*)d_in[27];
  const float* wv = (const float*)d_in[28]; const float* bv = (const float*)d_in[29];
  const float* wo = (const float*)d_in[30]; const float* bo = (const float*)d_in[31];
  const float* w1 = (const float*)d_in[32]; const float* b1 = (const float*)d_in[33];
  const float* w2 = (const float*)d_in[34]; const float* b2 = (const float*)d_in[35];
  float* out = (float*)d_out;

  // workspace layout (floats)
  const size_t SZ_BTD  = (size_t)B_ * T_ * D_;     // 4.19M
  const size_t SZ_BTE  = (size_t)B_ * T_ * ED_;    // 8.39M
  float* ws = (float*)d_ws;
  size_t o = 0;
  float* mod   = ws + o; o += 16384;
  float* xcur  = ws + o; o += SZ_BTD;
  float* bufA  = ws + o; o += SZ_BTD;   // xn / reused
  float* bufB  = ws + o; o += SZ_BTD;   // y_fwd / o-proj out / ffn out
  float* bufC  = ws + o; o += SZ_BTD;   // y_bwd raw / merged
  float* bufD  = ws + o; o += SZ_BTD;   // flipped input / y_bwd final
  float* xz    = ws + o; o += 2 * SZ_BTE;  // also S / FFN hidden
  float* xconv = ws + o; o += SZ_BTE;      // also q
  float* delta = ws + o; o += SZ_BTE;      // also attn raw out
  float* dbc   = ws + o; o += (size_t)B_ * T_ * (DR_ + 2 * NST_);
  float* ysb   = ws + o; o += SZ_BTE;
  float* kbuf  = ws + o; o += (size_t)B_ * TC_ * D_;
  float* vbuf  = ws + o; o += (size_t)B_ * TC_ * D_;

  const int M = B_ * T_;
  const int MC = B_ * TC_;

  // 1. AdaLN modulation
  k_modulation<<<(B_ * 9 * D_ + 255) / 256, 256, 0, stream>>>(t_emb, ada_w, ada_b, mod);

  // 2. xn = modulate(ln(x), c0, c1)
  k_ln_mod<<<M, 256, 0, stream>>>(x_in, mod, 0, 1, bufA);

  // 3. bidirectional Mamba
  run_mamba(stream, bufA, pf, xz, xconv, dbc, delta, ysb, bufB);           // forward -> bufB
  k_flip<<<(SZ_BTD + 255) / 256, 256, 0, stream>>>(bufA, bufD);            // flip input
  run_mamba(stream, bufD, pb, xz, xconv, dbc, delta, ysb, bufC);           // backward -> bufC
  k_flip<<<(SZ_BTD + 255) / 256, 256, 0, stream>>>(bufC, bufD);            // flip back -> bufD

  // merged = [y_fwd, y_bwd] @ merge_w  (two accumulating GEMMs) -> bufC
  gemm(stream, bufB, D_, merge_w,                  D_, 1, nullptr, bufC, D_, M, D_, D_, 0, 0);
  gemm(stream, bufD, D_, merge_w + (size_t)D_ * D_, D_, 1, nullptr, bufC, D_, M, D_, D_, 0, 1);

  // x = x + c2 * merged
  k_resgate<<<(SZ_BTD + 255) / 256, 256, 0, stream>>>(x_in, bufC, mod, 2, xcur);

  // 4. cross-attention
  k_ln_mod<<<M, 256, 0, stream>>>(xcur, mod, 3, 4, bufA);
  float* q = xconv;                                           // reuse (16MB <= 32MB)
  gemm(stream, bufA, D_, wq, D_, 1, bq, q, D_, M, D_, D_, 0, 0);
  gemm(stream, ctx, D_, wk, D_, 1, bk, kbuf, D_, MC, D_, D_, 0, 0);
  gemm(stream, ctx, D_, wv, D_, 1, bv, vbuf, D_, MC, D_, D_, 0, 0);
  k_rope<<<((size_t)B_ * T_ * NH_ * 32 + 255) / 256, 256, 0, stream>>>(q, T_);
  k_rope<<<((size_t)B_ * TC_ * NH_ * 32 + 255) / 256, 256, 0, stream>>>(kbuf, TC_);

  float* S = xz;          // T*TC floats, reused region
  float* attnraw = delta; // B*T*D, reused region
  for (int b = 0; b < B_; ++b) {
    for (int h = 0; h < NH_; ++h) {
      const float* qh = q    + (size_t)b * T_ * D_  + h * HD_;
      const float* kh = kbuf + (size_t)b * TC_ * D_ + h * HD_;
      const float* vh = vbuf + (size_t)b * TC_ * D_ + h * HD_;
      float* oh = attnraw + (size_t)b * T_ * D_ + h * HD_;
      gemm(stream, qh, D_, kh, 1, D_, nullptr, S, TC_, T_, TC_, HD_, 0, 0);   // S = q @ k^T
      k_softmax<<<T_, 256, 0, stream>>>(S, 0.125f);                           // /sqrt(64)
      gemm(stream, S, TC_, vh, D_, 1, nullptr, oh, D_, T_, HD_, TC_, 0, 0);   // O = P @ v
    }
  }
  gemm(stream, attnraw, D_, wo, D_, 1, bo, bufB, D_, M, D_, D_, 0, 0);
  k_resgate<<<(SZ_BTD + 255) / 256, 256, 0, stream>>>(xcur, bufB, mod, 5, xcur);

  // 5. FFN
  k_ln_mod<<<M, 256, 0, stream>>>(xcur, mod, 6, 7, bufA);
  float* hbuf = xz;  // B*T*DFF floats (64MB region)
  gemm(stream, bufA, D_, w1, DFF_, 1, b1, hbuf, DFF_, M, DFF_, D_, 2, 0);   // exact gelu
  gemm(stream, hbuf, DFF_, w2, D_, 1, b2, bufB, D_, M, D_, DFF_, 0, 0);
  k_resgate<<<(SZ_BTD + 255) / 256, 256, 0, stream>>>(xcur, bufB, mod, 8, out);
}
